// MODEL_23682449670855
// MI455X (gfx1250) — compile-verified, matrix operand order checked by reading, not run on hardware
//
#include <hip/hip_runtime.h>
#include <hip/hip_bf16.h>

// CDNA5 (gfx1250) wave32 WMMA implementation of the path-GRU scorer.
// Transposed formulation: gi^T = W_ih * x^T, gh^T = W_hh * h^T so the GRU
// state h stays layout-stationary. Weight fragments live in LDS (pre-packed
// per-lane) and are fetched per step with asm ds_load_b128 batches so they
// occupy only transient registers -- keeps the kernel under 256 VGPRs with
// zero scratch spills at waves_per_eu(4).
typedef __attribute__((ext_vector_type(16))) __bf16  bf16x16;
typedef __attribute__((ext_vector_type(8)))  float   f32x8;
typedef __attribute__((ext_vector_type(4)))  unsigned u32x4;

#define DIMD     32
#define PLEN     5
#define NREL     64
#define PAD_IDX  100000

__device__ __forceinline__ float fast_tanh(float x) {
#if __has_builtin(__builtin_amdgcn_tanhf)
    return __builtin_amdgcn_tanhf(x);      // gfx1250 native v_tanh_f32
#else
    return tanhf(x);
#endif
}

__device__ __forceinline__ float fast_sigmoid(float x) {
#if __has_builtin(__builtin_amdgcn_tanhf)
    // sigmoid(x) = 0.5*tanh(x/2) + 0.5 : mul + v_tanh + fma (1 trans op)
    return __builtin_amdgcn_tanhf(0.5f * x) * 0.5f + 0.5f;
#else
    return 1.0f / (1.0f + __expf(-x));
#endif
}

// Pack one 32-float row into the per-lane WMMA A/B fragment layout:
// lane-half hf holds K = {8hf..8hf+7} in elements 0..7 and K = {16+8hf..+7}
// in elements 8..15 (ISA 16-bit A layout; B col-major-per-lane is identical).
__device__ __forceinline__ bf16x16 pack_row_frag(const float* __restrict__ row, int hf) {
    const float4* r4 = (const float4*)row;           // rows are 128B-aligned
    float4 a0 = r4[2 * hf + 0];
    float4 a1 = r4[2 * hf + 1];
    float4 b0 = r4[4 + 2 * hf + 0];
    float4 b1 = r4[4 + 2 * hf + 1];
    bf16x16 o;
    o[0]  = (__bf16)a0.x; o[1]  = (__bf16)a0.y; o[2]  = (__bf16)a0.z; o[3]  = (__bf16)a0.w;
    o[4]  = (__bf16)a1.x; o[5]  = (__bf16)a1.y; o[6]  = (__bf16)a1.z; o[7]  = (__bf16)a1.w;
    o[8]  = (__bf16)b0.x; o[9]  = (__bf16)b0.y; o[10] = (__bf16)b0.z; o[11] = (__bf16)b0.w;
    o[12] = (__bf16)b1.x; o[13] = (__bf16)b1.y; o[14] = (__bf16)b1.z; o[15] = (__bf16)b1.w;
    return o;
}

// Fetch 6 pre-packed weight fragments (one per 16-row tile, this lane's slice)
// from LDS in a single asm batch: 12x ds_load_b128 + one s_wait_dscnt.
// asm volatile => opaque to LICM, so fragments stay transient (no hoist/spill).
__device__ __forceinline__ void lds_frag6(unsigned base, bf16x16 fr[6]) {
    u32x4 a0, a1, b0, b1, c0, c1, d0, d1, e0, e1, f0, f1;
    asm volatile(
        "ds_load_b128 %0, %12\n\t"
        "ds_load_b128 %1, %12 offset:16\n\t"
        "ds_load_b128 %2, %12 offset:1024\n\t"
        "ds_load_b128 %3, %12 offset:1040\n\t"
        "ds_load_b128 %4, %12 offset:2048\n\t"
        "ds_load_b128 %5, %12 offset:2064\n\t"
        "ds_load_b128 %6, %12 offset:3072\n\t"
        "ds_load_b128 %7, %12 offset:3088\n\t"
        "ds_load_b128 %8, %12 offset:4096\n\t"
        "ds_load_b128 %9, %12 offset:4112\n\t"
        "ds_load_b128 %10, %12 offset:5120\n\t"
        "ds_load_b128 %11, %12 offset:5136\n\t"
        "s_wait_dscnt 0"
        : "=&v"(a0), "=&v"(a1), "=&v"(b0), "=&v"(b1), "=&v"(c0), "=&v"(c1),
          "=&v"(d0), "=&v"(d1), "=&v"(e0), "=&v"(e1), "=&v"(f0), "=&v"(f1)
        : "v"(base));
    union { u32x4 u[2]; bf16x16 b; } cv;
    cv.u[0] = a0; cv.u[1] = a1; fr[0] = cv.b;
    cv.u[0] = b0; cv.u[1] = b1; fr[1] = cv.b;
    cv.u[0] = c0; cv.u[1] = c1; fr[2] = cv.b;
    cv.u[0] = d0; cv.u[1] = d1; fr[3] = cv.b;
    cv.u[0] = e0; cv.u[1] = e1; fr[4] = cv.b;
    cv.u[0] = f0; cv.u[1] = f1; fr[5] = cv.b;
}

__global__ void zero_f32_kernel(float* __restrict__ out, int n) {
    int i = blockIdx.x * blockDim.x + threadIdx.x;
    if (i < n) out[i] = 0.0f;
}

#define WMMA_BF16(A, B, C) __builtin_amdgcn_wmma_f32_16x16x32_bf16( \
        false, (A), false, (B), (short)0, (C), false, false)

__global__ void __launch_bounds__(256) __attribute__((amdgpu_waves_per_eu(4)))
gru_path_score_kernel(const int*   __restrict__ path,
                      const int*   __restrict__ path_idx,
                      const float* __restrict__ all_emb,
                      const float* __restrict__ edge_emb,
                      const float* __restrict__ virtual_emb,
                      const float* __restrict__ w_ih,
                      const float* __restrict__ w_hh,
                      const float* __restrict__ b_ih,
                      const float* __restrict__ b_hh,
                      const float* __restrict__ w_lin,
                      const float* __restrict__ b_lin,
                      float*       __restrict__ score,
                      int n_paths) {
    // 12 tiles x 32 lane-slots x 32B pre-packed bf16 fragments = 12 KB.
    __shared__ __align__(16) unsigned s_wfrag[384 * 8];

    const int tid  = threadIdx.x;
    const int lane = tid & 31;
    const int hf   = lane >> 4;     // K-half owned by this lane
    const int m    = lane & 15;     // weight-row / path slot within the tile

    // Stage per-lane weight fragments into LDS (once per block).
    for (int f = tid; f < 384; f += blockDim.x) {
        const int t  = f >> 5;      // 0..5 = W_ih tiles, 6..11 = W_hh tiles
        const int ls = f & 31;
        const int lm = ls & 15, lh = ls >> 4;
        const float* wsrc = (t < 6) ? (w_ih + (size_t)(16 * t + lm) * DIMD)
                                    : (w_hh + (size_t)(16 * (t - 6) + lm) * DIMD);
        union { bf16x16 b; unsigned u[8]; } cv;
        cv.b = pack_row_frag(wsrc, lh);
#pragma unroll
        for (int k = 0; k < 8; ++k) s_wfrag[f * 8 + k] = cv.u[k];
    }
    __syncthreads();

    // LDS byte offset of this lane's fragment slice (tile t at +t*1024).
    const unsigned lds_base = (unsigned)(size_t)(&s_wfrag[0]) + (unsigned)lane * 32u;

    const int wave    = blockIdx.x * (blockDim.x >> 5) + (tid >> 5);
    const int n_waves = gridDim.x * (blockDim.x >> 5);
    const int n_tiles = (n_paths + 15) >> 4;

    // Resident bias C-vectors. r/z gates get b_ih+b_hh combined (chained-WMMA
    // C); n gate keeps them split (r multiplies h_n only).
    f32x8 brz_c[4], bin_c[2], bhn_c[2];
#pragma unroll
    for (int t = 0; t < 4; ++t)
#pragma unroll
        for (int q = 0; q < 8; ++q)
            brz_c[t][q] = b_ih[16 * t + 8 * hf + q] + b_hh[16 * t + 8 * hf + q];
#pragma unroll
    for (int t = 0; t < 2; ++t)
#pragma unroll
        for (int q = 0; q < 8; ++q) {
            bin_c[t][q] = b_ih[64 + 16 * t + 8 * hf + q];
            bhn_c[t][q] = b_hh[64 + 16 * t + 8 * hf + q];
        }

    float wl[16];
#pragma unroll
    for (int q = 0; q < 8; ++q) {
        wl[q]     = w_lin[8 * hf + q];
        wl[8 + q] = w_lin[16 + 8 * hf + q];
    }
    const float blin = b_lin[0];

    for (int tile = wave; tile < n_tiles; tile += n_waves) {
        const int  p     = tile * 16 + m;
        const bool valid = (p < n_paths);
        const int  pc    = valid ? p : (n_paths - 1);

        // h in C-layout: hc[q] = h[p, 8hf+q], hc[8+q] = h[p, 16+8hf+q]
        float hc[16];
#pragma unroll
        for (int i = 0; i < 16; ++i) hc[i] = 0.0f;

#pragma unroll
        for (int l = 0; l < PLEN; ++l) {
            // Gather this path's step-l embedding row (tables are L2-resident).
            const int idx = path[pc * PLEN + l];
            const bool is_rel = ((l & 1) == 1) && (idx != PAD_IDX);
            const float* row = is_rel
                ? ((idx < NREL) ? (edge_emb + (size_t)idx * DIMD) : virtual_emb)
                : (all_emb + (size_t)idx * DIMD);
            const bf16x16 xf = pack_row_frag(row, hf);

            // h as B fragment: identical element ownership, in-lane cvt only.
            bf16x16 hb;
#pragma unroll
            for (int j = 0; j < 16; ++j) hb[j] = (__bf16)hc[j];

            bf16x16 wf[6];

            // Phase 1: input-side GEMMs, biases folded into C.
            lds_frag6(lds_base, wf);                 // W_ih tiles 0..5
            f32x8 gr0 = WMMA_BF16(wf[0], xf, brz_c[0]);
            f32x8 gr1 = WMMA_BF16(wf[1], xf, brz_c[1]);
            f32x8 gz0 = WMMA_BF16(wf[2], xf, brz_c[2]);
            f32x8 gz1 = WMMA_BF16(wf[3], xf, brz_c[3]);
            f32x8 gin0 = WMMA_BF16(wf[4], xf, bin_c[0]);
            f32x8 gin1 = WMMA_BF16(wf[5], xf, bin_c[1]);

            // Phase 2: hidden-side GEMMs; r/z chained in place via C operand.
            lds_frag6(lds_base + 6144u, wf);         // W_hh tiles 0..5
            gr0 = WMMA_BF16(wf[0], hb, gr0);
            gr1 = WMMA_BF16(wf[1], hb, gr1);
            gz0 = WMMA_BF16(wf[2], hb, gz0);
            gz1 = WMMA_BF16(wf[3], hb, gz1);
            f32x8 ghn0 = WMMA_BF16(wf[4], hb, bhn_c[0]);
            f32x8 ghn1 = WMMA_BF16(wf[5], hb, bhn_c[1]);

            // GRU elementwise update (native v_tanh_f32 path).
#pragma unroll
            for (int q = 0; q < 8; ++q) {
                const float r0 = fast_sigmoid(gr0[q]);
                const float z0 = fast_sigmoid(gz0[q]);
                const float n0 = fast_tanh(gin0[q] + r0 * ghn0[q]);
                hc[q] = (1.0f - z0) * n0 + z0 * hc[q];
                const float r1 = fast_sigmoid(gr1[q]);
                const float z1 = fast_sigmoid(gz1[q]);
                const float n1 = fast_tanh(gin1[q] + r1 * ghn1[q]);
                hc[8 + q] = (1.0f - z1) * n1 + z1 * hc[8 + q];
            }
        }

        // per_path = h . w_lin + b_lin ; merge the two K-halves across lane^16.
        float part = 0.0f;
#pragma unroll
        for (int j = 0; j < 16; ++j) part += hc[j] * wl[j];
        part += __shfl_xor(part, 16);
        if (hf == 0 && valid) {
            atomicAdd(score + path_idx[p], part + blin);
        }
    }
}

extern "C" void kernel_launch(void* const* d_in, const int* in_sizes, int n_in,
                              void* d_out, int out_size, void* d_ws, size_t ws_size,
                              hipStream_t stream) {
    // setup_inputs order:
    // 0 users(i32,unused) 1 path(i32) 2 path_idx(i32) 3 all_emb 4 edge_emb
    // 5 virtual_emb 6 w_ih 7 w_hh 8 b_ih 9 b_hh 10 w_lin 11 b_lin
    const int*   path        = (const int*)d_in[1];
    const int*   path_idx    = (const int*)d_in[2];
    const float* all_emb     = (const float*)d_in[3];
    const float* edge_emb    = (const float*)d_in[4];
    const float* virtual_emb = (const float*)d_in[5];
    const float* w_ih        = (const float*)d_in[6];
    const float* w_hh        = (const float*)d_in[7];
    const float* b_ih        = (const float*)d_in[8];
    const float* b_hh        = (const float*)d_in[9];
    const float* w_lin       = (const float*)d_in[10];
    const float* b_lin       = (const float*)d_in[11];
    float* score = (float*)d_out;
    const int n_paths = in_sizes[2];   // N_PATHS

    zero_f32_kernel<<<(out_size + 255) / 256, 256, 0, stream>>>(score, out_size);

    const int threads = 256;           // 8 wave32 per block
    const int blocks  = 512;           // grid-stride over 18750 16-path tiles
    gru_path_score_kernel<<<blocks, threads, 0, stream>>>(
        path, path_idx, all_emb, edge_emb, virtual_emb,
        w_ih, w_hh, b_ih, b_hh, w_lin, b_lin, score, n_paths);
}
